// ConvTokenEmbedder_16527034155751
// MI455X (gfx1250) — compile-verified, matrix operand order checked by reading, not run on hardware
//
#include <hip/hip_runtime.h>
#include <hip/hip_bf16.h>

typedef __bf16 bf16_t;
typedef __bf16 v16bf __attribute__((ext_vector_type(16)));
typedef __bf16 v8bf  __attribute__((ext_vector_type(8)));
typedef float  v8f   __attribute__((ext_vector_type(8)));

#define NTOK 4096
#define NF   2048
#define XROW 832    // padded K-row per token (real length 800 = 50*16)
#define LDSP 40     // LDS A-tile row pitch (elements): 16 distinct bank groups

// ---------------------------------------------------------------------------
// helpers
// ---------------------------------------------------------------------------
__device__ __forceinline__ v8f wmma_bf16(v16bf a, v16bf b, v8f c) {
  // D = A(16x32) * B(32x16) + C, f32 accumulate
  return __builtin_amdgcn_wmma_f32_16x16x32_bf16(false, a, false, b, (short)0, c,
                                                 false, false);
}

// A/B fragment load from a row-major bf16 row: per-lane K chunks at
// base + (lane>>4)*8 and base + (lane>>4)*8 + 16  (ISA 16-bit 16x32 layout).
// Works for both global and LDS pointers (address space inferred after inline).
__device__ __forceinline__ v16bf load_frag_bf16(const bf16_t* row, int off) {
  v8bf lo = *(const v8bf*)(row + off);
  v8bf hi = *(const v8bf*)(row + off + 16);
  v16bf r;
#pragma unroll
  for (int i = 0; i < 8; ++i) { r[i] = lo[i]; r[i + 8] = hi[i]; }
  return r;
}

__device__ __forceinline__ v8f zero8() {
  v8f z;
#pragma unroll
  for (int i = 0; i < 8; ++i) z[i] = 0.f;
  return z;
}

// ---------------------------------------------------------------------------
// 1) embed: gather char embeddings, fp32 -> bf16, token-major K rows (padded)
// ---------------------------------------------------------------------------
__global__ void embed_kernel(const int* __restrict__ chars,
                             const float* __restrict__ emb,
                             bf16_t* __restrict__ x) {
  int tid = blockIdx.x * blockDim.x + threadIdx.x;
  const int total = NTOK * XROW;
  if (tid >= total) return;
  int tok = tid / XROW;
  int e   = tid - tok * XROW;
  float v = 0.f;
  if (e < 800) {
    int j = e >> 4;          // char position
    int c = e & 15;          // embedding channel
    int ci = chars[tok * 50 + j];
    v = emb[ci * 16 + c];
  }
  x[tid] = (bf16_t)v;
}

// ---------------------------------------------------------------------------
// 2a) pack conv weights: (n, 16, w) fp32 -> (n, Kp) bf16, K index = t*16+c
// ---------------------------------------------------------------------------
__global__ void pack_conv_w(const float* __restrict__ w, bf16_t* __restrict__ dst,
                            int nf, int width, int Kp) {
  int tid = blockIdx.x * blockDim.x + threadIdx.x;
  int total = nf * Kp;
  if (tid >= total) return;
  int f = tid / Kp;
  int k = tid - f * Kp;
  float v = 0.f;
  if (k < 16 * width) {
    int t = k >> 4;
    int c = k & 15;
    v = w[(f * 16 + c) * width + t];   // layout (n, CHAR_DIM, w)
  }
  dst[tid] = (bf16_t)v;
}

// ---------------------------------------------------------------------------
// 2b) straight fp32 -> bf16 pack (highway / projection weights, one-time)
// ---------------------------------------------------------------------------
__global__ void pack_f32_bf16(const float* __restrict__ src,
                              bf16_t* __restrict__ dst, int total) {
  int tid = blockIdx.x * blockDim.x + threadIdx.x;
  if (tid >= total) return;
  dst[tid] = (bf16_t)src[tid];
}

// ---------------------------------------------------------------------------
// 3) conv as GEMM + max-over-time + ReLU, one wave = 16 tokens x 16 filters
//    WIDTH is a template parameter -> branch-free fully unrolled inner loop
// ---------------------------------------------------------------------------
template <int WIDTH>
__global__ void conv_gemm_kernel(const bf16_t* __restrict__ x,
                                 const bf16_t* __restrict__ wbf,
                                 const float* __restrict__ bias,
                                 bf16_t* __restrict__ h,
                                 int colOff) {
  constexpr int NK   = (16 * WIDTH + 31) / 32;  // K chunks of 32
  constexpr int KP   = NK * 32;                 // padded K
  constexpr int NPOS = 51 - WIDTH;              // conv output positions

  const int lane = threadIdx.x & 31;
  const int n    = lane & 15;    // N (filter) lane index
  const int kh   = lane >> 4;    // K-half selector
  const int tm   = blockIdx.x;   // token tile
  const int fn   = blockIdx.y;   // filter tile within group

  const bf16_t* xrow = x + (size_t)(tm * 16 + n) * XROW;
  const bf16_t* wrow = wbf + (size_t)(fn * 16 + n) * KP;

  // B fragments are time-invariant: preload
  v16bf bfrag[NK];
#pragma unroll
  for (int c = 0; c < NK; ++c) bfrag[c] = load_frag_bf16(wrow, c * 32 + kh * 8);

  v8f maxv;
#pragma unroll
  for (int i = 0; i < 8; ++i) maxv[i] = -1e30f;

#pragma unroll 2
  for (int t = 0; t < NPOS; ++t) {
    v8f acc = zero8();
#pragma unroll
    for (int c = 0; c < NK; ++c) {
      v16bf a = load_frag_bf16(xrow, 16 * t + c * 32 + kh * 8);
      acc = wmma_bf16(a, bfrag[c], acc);
    }
#pragma unroll
    for (int r = 0; r < 8; ++r) maxv[r] = fmaxf(maxv[r], acc[r]);
  }

  float bv  = bias[fn * 16 + n];
  int   col = colOff + fn * 16 + n;
#pragma unroll
  for (int r = 0; r < 8; ++r) {
    int row = tm * 16 + kh * 8 + r;           // D layout: M = 8*(lane>>4)+r
    float v = maxv[r] + bv;
    v = v > 0.f ? v : 0.f;
    h[(size_t)row * NF + col] = (bf16_t)v;
  }
}

// ---------------------------------------------------------------------------
// 4) highway: p = h @ W^T + b ; h = sig(p_g) * h + (1-sig(p_g)) * relu(p_t)
//    block = 4 waves sharing one 64-token A-tile staged in LDS (double
//    buffered); each wave owns one 16-feature column tile and accumulates
//    both transform (j) and gate (j+2048) -> 8 WMMAs per K-step per wave.
//    All 4 A-fragments are hoisted into registers before the WMMA burst so
//    one ds-wait covers 8 matrix ops.
// ---------------------------------------------------------------------------
__global__ void highway_kernel(const bf16_t* __restrict__ hin,
                               const bf16_t* __restrict__ Wb,   // packed bf16
                               const float* __restrict__ b,
                               bf16_t* __restrict__ hout) {
  __shared__ bf16_t sA[2][64 * LDSP];

  const int tid  = threadIdx.x;
  const int lane = tid & 31;
  const int wid  = tid >> 5;
  const int n    = lane & 15;
  const int kh   = lane >> 4;
  const int m0   = blockIdx.x * 64;
  const int j0   = (blockIdx.y * 4 + wid) * 16;

  // cooperative A-stage mapping: 128 threads x 16 elems = 64 rows x 32 cols
  const int lr = tid >> 1;          // A row 0..63
  const int lc = (tid & 1) * 16;    // col 0 or 16
  const bf16_t* gA = hin + (size_t)(m0 + lr) * NF + lc;

  const bf16_t* wt = Wb + (size_t)(j0 + n) * NF;          // transform rows
  const bf16_t* wg = Wb + (size_t)(j0 + NF + n) * NF;     // gate rows

  v8f at[4], ag[4];
#pragma unroll
  for (int mt = 0; mt < 4; ++mt) { at[mt] = zero8(); ag[mt] = zero8(); }

  // prologue: stage first K-slab
  {
    v8bf a0 = *(const v8bf*)(gA);
    v8bf a1 = *(const v8bf*)(gA + 8);
    *(v8bf*)(&sA[0][lr * LDSP + lc])     = a0;
    *(v8bf*)(&sA[0][lr * LDSP + lc + 8]) = a1;
  }
  __syncthreads();

  for (int kb = 0; kb < NF; kb += 32) {
    const int p = (kb >> 5) & 1;
    if (kb + 32 < NF) {   // stage next slab into the other buffer
      v8bf a0 = *(const v8bf*)(gA + kb + 32);
      v8bf a1 = *(const v8bf*)(gA + kb + 40);
      *(v8bf*)(&sA[p ^ 1][lr * LDSP + lc])     = a0;
      *(v8bf*)(&sA[p ^ 1][lr * LDSP + lc + 8]) = a1;
    }
    v16bf bt = load_frag_bf16(wt, kb + kh * 8);
    v16bf bg = load_frag_bf16(wg, kb + kh * 8);
    v16bf afr[4];
#pragma unroll
    for (int mt = 0; mt < 4; ++mt)
      afr[mt] = load_frag_bf16(&sA[p][(mt * 16 + n) * LDSP], kh * 8);
#pragma unroll
    for (int mt = 0; mt < 4; ++mt) {
      at[mt] = wmma_bf16(afr[mt], bt, at[mt]);
      ag[mt] = wmma_bf16(afr[mt], bg, ag[mt]);
    }
    __syncthreads();
  }

  int   col  = j0 + n;
  float bt_b = b[col];
  float bg_b = b[NF + col];
#pragma unroll
  for (int mt = 0; mt < 4; ++mt) {
#pragma unroll
    for (int r = 0; r < 8; ++r) {
      int row = m0 + mt * 16 + kh * 8 + r;
      float tv = at[mt][r] + bt_b;
      tv = tv > 0.f ? tv : 0.f;
      float gv = 1.f / (1.f + __expf(-(ag[mt][r] + bg_b)));
      float ho = (float)hin[(size_t)row * NF + col];
      hout[(size_t)row * NF + col] = (bf16_t)(gv * ho + (1.f - gv) * tv);
    }
  }
}

// ---------------------------------------------------------------------------
// 5) projection: out = h @ proj_w^T + proj_b (fp32 out), same LDS staging
// ---------------------------------------------------------------------------
__global__ void proj_kernel(const bf16_t* __restrict__ hin,
                            const bf16_t* __restrict__ Wb,   // packed bf16
                            const float* __restrict__ b,
                            float* __restrict__ out) {
  __shared__ bf16_t sA[2][64 * LDSP];

  const int tid  = threadIdx.x;
  const int lane = tid & 31;
  const int wid  = tid >> 5;
  const int n    = lane & 15;
  const int kh   = lane >> 4;
  const int m0   = blockIdx.x * 64;
  const int n0   = (blockIdx.y * 4 + wid) * 16;

  const int lr = tid >> 1;
  const int lc = (tid & 1) * 16;
  const bf16_t* gA = hin + (size_t)(m0 + lr) * NF + lc;
  const bf16_t* wr = Wb + (size_t)(n0 + n) * NF;

  v8f acc[4];
#pragma unroll
  for (int mt = 0; mt < 4; ++mt) acc[mt] = zero8();

  {
    v8bf a0 = *(const v8bf*)(gA);
    v8bf a1 = *(const v8bf*)(gA + 8);
    *(v8bf*)(&sA[0][lr * LDSP + lc])     = a0;
    *(v8bf*)(&sA[0][lr * LDSP + lc + 8]) = a1;
  }
  __syncthreads();

  for (int kb = 0; kb < NF; kb += 32) {
    const int p = (kb >> 5) & 1;
    if (kb + 32 < NF) {
      v8bf a0 = *(const v8bf*)(gA + kb + 32);
      v8bf a1 = *(const v8bf*)(gA + kb + 40);
      *(v8bf*)(&sA[p ^ 1][lr * LDSP + lc])     = a0;
      *(v8bf*)(&sA[p ^ 1][lr * LDSP + lc + 8]) = a1;
    }
    v16bf bf = load_frag_bf16(wr, kb + kh * 8);
    v16bf afr[4];
#pragma unroll
    for (int mt = 0; mt < 4; ++mt)
      afr[mt] = load_frag_bf16(&sA[p][(mt * 16 + n) * LDSP], kh * 8);
#pragma unroll
    for (int mt = 0; mt < 4; ++mt)
      acc[mt] = wmma_bf16(afr[mt], bf, acc[mt]);
    __syncthreads();
  }

  int   col = n0 + n;
  float bb  = b[col];
#pragma unroll
  for (int mt = 0; mt < 4; ++mt) {
#pragma unroll
    for (int r = 0; r < 8; ++r) {
      int row = m0 + mt * 16 + kh * 8 + r;
      out[(size_t)row * 512 + col] = acc[mt][r] + bb;
    }
  }
}

// ---------------------------------------------------------------------------
// launch
// ---------------------------------------------------------------------------
extern "C" void kernel_launch(void* const* d_in, const int* in_sizes, int n_in,
                              void* d_out, int out_size, void* d_ws, size_t ws_size,
                              hipStream_t stream) {
  (void)in_sizes; (void)n_in; (void)out_size; (void)ws_size;

  const int*   chars = (const int*)d_in[1];
  const float* emb   = (const float*)d_in[2];

  // workspace layout (bf16 elements)
  char* ws = (char*)d_ws;
  bf16_t* x   = (bf16_t*)ws;                                   // 4096*832
  bf16_t* h0  = (bf16_t*)(ws + (size_t)NTOK * XROW * 2);       // 4096*2048
  bf16_t* h1  = (bf16_t*)((char*)h0 + (size_t)NTOK * NF * 2);  // 4096*2048
  bf16_t* cw  = (bf16_t*)((char*)h1 + (size_t)NTOK * NF * 2);  // packed conv w
  const size_t cwTotal = 32 * 32 + 32 * 32 + 64 * 64 + 128 * 64 +
                         256 * 96 + 512 * 96 + 1024 * 128;
  bf16_t* wb0 = cw + cwTotal;                                  // 2*NF*NF
  bf16_t* wb1 = wb0 + (size_t)2 * NF * NF;                     // 2*NF*NF
  bf16_t* pwb = wb1 + (size_t)2 * NF * NF;                     // 512*NF

  // 1) embed + bf16 convert
  {
    int total = NTOK * XROW;
    embed_kernel<<<(total + 255) / 256, 256, 0, stream>>>(chars, emb, x);
  }

  // conv group descriptors
  const int nfs[7]    = {32, 32, 64, 128, 256, 512, 1024};
  const int kps[7]    = {32, 32, 64, 64, 96, 96, 128};
  const int widths[7] = {1, 2, 3, 4, 5, 6, 7};
  const int colOff[7] = {0, 32, 64, 128, 256, 512, 1024};

  // 2a) pack conv weights to bf16 (zero-padded K)
  size_t cwOff[7];
  {
    size_t o = 0;
    for (int g = 0; g < 7; ++g) {
      cwOff[g] = o;
      o += (size_t)nfs[g] * kps[g];
      int total = nfs[g] * kps[g];
      pack_conv_w<<<(total + 255) / 256, 256, 0, stream>>>(
          (const float*)d_in[3 + 2 * g], cw + cwOff[g], nfs[g], widths[g], kps[g]);
    }
  }

  // 2b) pack highway / projection weights to bf16 (one-time)
  {
    int tHW = 2 * NF * NF;     // 8.39M
    pack_f32_bf16<<<(tHW + 255) / 256, 256, 0, stream>>>(
        (const float*)d_in[17], wb0, tHW);
    pack_f32_bf16<<<(tHW + 255) / 256, 256, 0, stream>>>(
        (const float*)d_in[19], wb1, tHW);
    int tPR = 512 * NF;        // 1.05M
    pack_f32_bf16<<<(tPR + 255) / 256, 256, 0, stream>>>(
        (const float*)d_in[21], pwb, tPR);
  }

  // 3) conv GEMM + maxpool + relu -> h0 (templated width -> branch-free)
  {
#define LAUNCH_CONV(G, WID)                                                 \
    conv_gemm_kernel<WID><<<dim3(NTOK / 16, nfs[G] / 16), 32, 0, stream>>>( \
        x, cw + cwOff[G], (const float*)d_in[4 + 2 * (G)], h0, colOff[G]);
    LAUNCH_CONV(0, 1)
    LAUNCH_CONV(1, 2)
    LAUNCH_CONV(2, 3)
    LAUNCH_CONV(3, 4)
    LAUNCH_CONV(4, 5)
    LAUNCH_CONV(5, 6)
    LAUNCH_CONV(6, 7)
#undef LAUNCH_CONV
  }

  // 4) highway x2 (ping-pong h0 -> h1 -> h0)
  {
    dim3 grid(NTOK / 64, NF / (16 * 4));
    highway_kernel<<<grid, 128, 0, stream>>>(h0, wb0, (const float*)d_in[18], h1);
    highway_kernel<<<grid, 128, 0, stream>>>(h1, wb1, (const float*)d_in[20], h0);
  }

  // 5) projection -> d_out
  {
    dim3 grid(NTOK / 64, 512 / (16 * 4));
    proj_kernel<<<grid, 128, 0, stream>>>(h0, pwb, (const float*)d_in[22],
                                          (float*)d_out);
  }
}
